// ResidualVectorQuantizer_88012469829945
// MI455X (gfx1250) — compile-verified, matrix operand order checked by reading, not run on hardware
//
#include <hip/hip_runtime.h>
#include <hip/hip_bf16.h>

#define NUM_LEVELS 8
#define CODEBOOK   1024
#define DIM        256
#define BATCH      32768
#define BETA       0.25f

typedef __attribute__((ext_vector_type(16))) __bf16       bf16x16;
typedef __attribute__((ext_vector_type(8)))  float        f32x8;
typedef __attribute__((ext_vector_type(4)))  unsigned int u32x4;
typedef __attribute__((ext_vector_type(8)))  int          i32x8;
typedef __attribute__((ext_vector_type(4)))  int          i32x4;

// LDS tile: 16 codebook rows x 256 K bf16, padded 512B->528B per row (bank spread)
#define ROW_BYTES   528
#define TILE_BYTES  (16 * ROW_BYTES)

// ---------------- workspace layout (bytes) ----------------
static constexpr size_t RES_F32_OFF  = 0;                                            // 32 MB
static constexpr size_t RES_BF16_OFF = RES_F32_OFF  + (size_t)BATCH * DIM * 4;       // +16 MB
static constexpr size_t EMB_BF16_OFF = RES_BF16_OFF + (size_t)BATCH * DIM * 2;       // +4 MB
static constexpr size_t NORMS_OFF    = EMB_BF16_OFF + (size_t)NUM_LEVELS * CODEBOOK * DIM * 2;
static constexpr size_t IDX_OFF      = NORMS_OFF    + (size_t)NUM_LEVELS * CODEBOOK * 4;
static constexpr size_t COUNTS_OFF   = IDX_OFF      + (size_t)BATCH * 4;
static constexpr size_t COMMIT_OFF   = COUNTS_OFF   + (size_t)CODEBOOK * 4;

// ---------------- TDM: one 16x256 bf16 tile -> LDS, padded rows ----------------
// D# group1 word0: data_size=2B (bits17:16=1), pad_enable (bit20),
// pad_interval=128 DWORDs (bits24:22=6), pad_amount=4 DWORDs (bits31:25=3).
__device__ inline void tdm_load_tile(const __bf16* gsrc, unsigned ldsAddr) {
  unsigned long long ga = (unsigned long long)(uintptr_t)gsrc;
  u32x4 g0 = { 1u,                                  // count=1, user descriptor
               ldsAddr,                             // LDS byte address
               (unsigned)(ga & 0xffffffffu),        // global_addr[31:0]
               (unsigned)((ga >> 32) & 0x1ffffffu)  // global_addr[56:32]
                 | (2u << 30) };                    // type=2 ("image")
  i32x8 g1 = { (int)((1u << 16) | (1u << 20) | (6u << 22) | (3u << 25)),
               (int)(256u << 16),    // tensor_dim0 = 256 (lo16 @ [31:16])
               (int)(1024u << 16),   // tensor_dim1 = 1024 (lo16 @ [31:16])
               (int)(256u << 16),    // tile_dim0 = 256 (@ [31:16])
               16,                   // tile_dim1 = 16 rows
               256,                  // tensor_dim0_stride[31:0] = 256 elems
               0, 0 };
  i32x4 gz4 = {};
  i32x8 gz8 = {};
  __builtin_amdgcn_tensor_load_to_lds(g0, g1, gz4, gz4, gz8, 0);  // clang-23 6-arg form
}

// ---------------- init: residual = z (f32 + bf16), zq accumulator = 0 ----------------
__global__ __launch_bounds__(256) void rvq_init_residual(const float* __restrict__ z,
                                                         float* __restrict__ residual,
                                                         __bf16* __restrict__ resb,
                                                         float* __restrict__ zq) {
  size_t i = (size_t)blockIdx.x * blockDim.x + threadIdx.x;
  if (i < (size_t)BATCH * DIM) {
    float v = z[i];
    residual[i] = v;
    resb[i]     = (__bf16)v;
    zq[i]       = 0.0f;
  }
}

__global__ __launch_bounds__(1024) void rvq_init_scalars(int* __restrict__ counts,
                                                         float* __restrict__ commit) {
  counts[threadIdx.x] = 0;
  if (threadIdx.x == 0) *commit = 0.0f;
}

// ---------------- codebook prep: bf16 copy + row norms ----------------
__global__ __launch_bounds__(256) void rvq_emb_prep(const float* __restrict__ emb,
                                                    __bf16* __restrict__ embb,
                                                    float* __restrict__ norms) {
  const int row = blockIdx.x;           // flattened lvl*CODEBOOK + k
  const int d   = threadIdx.x;
  const size_t o = (size_t)row * DIM + d;
  float v = emb[o];
  embb[o] = (__bf16)v;
  __shared__ float s[256];
  s[d] = v * v;
  __syncthreads();
  for (int off = 128; off > 0; off >>= 1) {
    if (d < off) s[d] += s[d + off];
    __syncthreads();
  }
  if (d == 0) norms[row] = s[0];
}

// ---------------- fused distance GEMM (WMMA bf16, TDM-staged B) + argmin ----------------
// grid: BATCH/64 blocks of 128 threads. Block covers 64 rows; wave w owns rows
// [base+w*16, base+w*16+16) across ALL 1024 columns. B tiles staged to LDS by the
// Tensor Data Mover (wave 0), double-buffered, shared by the 4 waves.
__global__ __launch_bounds__(128) void rvq_argmin_kernel(const __bf16* __restrict__ resb,
                                                         const __bf16* __restrict__ embb,   // level slice
                                                         const float*  __restrict__ norms,  // level slice
                                                         int*   __restrict__ idx_ws,
                                                         float* __restrict__ idx_out,       // (B, L) row-major
                                                         int*   __restrict__ counts,
                                                         int lvl) {
  __shared__ __align__(16) unsigned char smemB[2][TILE_BYTES];

  const int wave = threadIdx.x >> 5;
  const int lane = threadIdx.x & 31;
  const int col  = lane & 15;
  const int hi   = lane >> 4;
  const int rowBase = blockIdx.x * 64 + wave * 16;

  // Preload A: this lane supplies row (rowBase+col); half hi holds k = kt*32 + 16*hi + s (contiguous), same order as B.
  bf16x16 a[8];
  const __bf16* arow = resb + (size_t)(rowBase + col) * DIM + hi * 16;
#pragma unroll
  for (int kt = 0; kt < 8; ++kt)
    a[kt] = *reinterpret_cast<const bf16x16*>(arow + kt * 32);

  float bestVal[8];
  int   bestIdx[8];
#pragma unroll
  for (int j = 0; j < 8; ++j) { bestVal[j] = 3.4e38f; bestIdx[j] = 0; }

  if (wave == 0)
    tdm_load_tile(embb, (unsigned)(uintptr_t)&smemB[0][0]);

  for (int nt = 0; nt < 64; ++nt) {
    if (wave == 0) {
      if (nt + 1 < 64) {
        tdm_load_tile(embb + (size_t)(nt + 1) * 16 * DIM,
                      (unsigned)(uintptr_t)&smemB[(nt + 1) & 1][0]);
        __builtin_amdgcn_s_wait_tensorcnt(1);   // tile nt has landed
      } else {
        __builtin_amdgcn_s_wait_tensorcnt(0);
      }
    }
    __syncthreads();                            // tile nt visible to all waves

    const unsigned char* bbuf = &smemB[nt & 1][0];
    const int ncol = nt * 16 + col;             // this lane's codebook column
    f32x8 acc = {};
#pragma unroll
    for (int kt = 0; kt < 8; ++kt) {
      bf16x16 b = *reinterpret_cast<const bf16x16*>(bbuf + col * ROW_BYTES + hi * 32 + kt * 64);
      acc = __builtin_amdgcn_wmma_f32_16x16x32_bf16(false, a[kt], false, b,
                                                    (short)0, acc, false, false);
    }
    const float en = norms[ncol];
#pragma unroll
    for (int j = 0; j < 8; ++j) {               // acc[j] -> row hi*8+j, column `ncol`
      float dist = en - 2.0f * acc[j];
      if (dist < bestVal[j]) { bestVal[j] = dist; bestIdx[j] = ncol; }
    }
    __syncthreads();                            // done reading buf before TDM reuses it
  }

  // Butterfly reduce across the 16 column-lanes (xor bits 0..3 stay within the half).
#pragma unroll
  for (int m = 1; m <= 8; m <<= 1) {
#pragma unroll
    for (int j = 0; j < 8; ++j) {
      float ov = __shfl_xor(bestVal[j], m, 32);
      int   oi = __shfl_xor(bestIdx[j], m, 32);
      if (ov < bestVal[j] || (ov == bestVal[j] && oi < bestIdx[j])) {
        bestVal[j] = ov; bestIdx[j] = oi;
      }
    }
  }

  if (col == 0) {
#pragma unroll
    for (int j = 0; j < 8; ++j) {
      const int grow = rowBase + hi * 8 + j;
      const int bi = bestIdx[j];
      idx_ws[grow] = bi;
      idx_out[(size_t)grow * NUM_LEVELS + lvl] = (float)bi;
      atomicAdd(&counts[bi], 1);
    }
  }
}

// ---------------- gather + residual/commit/qsum update ----------------
__global__ __launch_bounds__(256) void rvq_update_kernel(const float* __restrict__ emb,  // level slice f32
                                                         const int* __restrict__ idx_ws,
                                                         float* __restrict__ residual,
                                                         __bf16* __restrict__ resb,
                                                         float* __restrict__ qsum,
                                                         float* __restrict__ commit) {
  const int b = blockIdx.x;
  const int d = threadIdx.x;
  const int k = idx_ws[b];
  const float q = emb[(size_t)k * DIM + d];
  const size_t o = (size_t)b * DIM + d;
  const float r = residual[o];
  const float diff = q - r;
  const float nr = r - q;
  residual[o] = nr;
  resb[o]     = (__bf16)nr;
  qsum[o]    += q;

  __shared__ float s[256];
  s[d] = diff * diff;
  __syncthreads();
  for (int off = 128; off > 0; off >>= 1) {
    if (d < off) s[d] += s[d + off];
    __syncthreads();
  }
  if (d == 0) atomicAdd(commit, s[0]);
}

// ---------------- perplexity + scalars ----------------
__global__ __launch_bounds__(1024) void rvq_finalize_kernel(const int* __restrict__ counts,
                                                            const float* __restrict__ commit,
                                                            float* __restrict__ out_scalars) {
  const int t = threadIdx.x;
  __shared__ float s[1024];
  float c = (float)counts[t];
  float p = c / ((float)(BATCH * NUM_LEVELS) + 1e-10f);
  s[t] = (p > 0.0f) ? -p * __logf(p + 1e-10f) : 0.0f;
  __syncthreads();
  for (int off = 512; off > 0; off >>= 1) {
    if (t < off) s[t] += s[t + off];
    __syncthreads();
  }
  if (t == 0) {
    float entropy = s[0];
    float total_commit = *commit / (float)((size_t)BATCH * DIM);
    out_scalars[0] = BETA * total_commit;   // vq_loss
    out_scalars[1] = total_commit;          // total_commit
    out_scalars[2] = __expf(entropy);       // perplexity
  }
}

extern "C" void kernel_launch(void* const* d_in, const int* in_sizes, int n_in,
                              void* d_out, int out_size, void* d_ws, size_t ws_size,
                              hipStream_t stream) {
  const float* z   = (const float*)d_in[0];   // (BATCH, DIM)
  const float* emb = (const float*)d_in[1];   // (L, K, DIM)

  char* ws = (char*)d_ws;
  float*  residual = (float*) (ws + RES_F32_OFF);
  __bf16* resb     = (__bf16*)(ws + RES_BF16_OFF);
  __bf16* embb     = (__bf16*)(ws + EMB_BF16_OFF);
  float*  norms    = (float*) (ws + NORMS_OFF);
  int*    idx_ws   = (int*)   (ws + IDX_OFF);
  int*    counts   = (int*)   (ws + COUNTS_OFF);
  float*  commit   = (float*) (ws + COMMIT_OFF);

  float* zq      = (float*)d_out;                       // z_q == quantized_sum (value)
  float* idx_out = zq + (size_t)BATCH * DIM;            // (B, L) as floats
  float* scal    = idx_out + (size_t)BATCH * NUM_LEVELS;

  rvq_init_residual<<<(BATCH * DIM) / 256, 256, 0, stream>>>(z, residual, resb, zq);
  rvq_init_scalars<<<1, 1024, 0, stream>>>(counts, commit);
  rvq_emb_prep<<<NUM_LEVELS * CODEBOOK, 256, 0, stream>>>(emb, embb, norms);

  for (int lvl = 0; lvl < NUM_LEVELS; ++lvl) {
    rvq_argmin_kernel<<<BATCH / 64, 128, 0, stream>>>(
        resb,
        embb  + (size_t)lvl * CODEBOOK * DIM,
        norms + (size_t)lvl * CODEBOOK,
        idx_ws, idx_out, counts, lvl);
    rvq_update_kernel<<<BATCH, 256, 0, stream>>>(
        emb + (size_t)lvl * CODEBOOK * DIM,
        idx_ws, residual, resb, zq, commit);
  }

  rvq_finalize_kernel<<<1, 1024, 0, stream>>>(counts, commit, scal);
}
